// LatentAttention_71451075936554
// MI455X (gfx1250) — matrix operand order repaired and checked by reading, not compiled, not run on hardware
//
#include <hip/hip_runtime.h>
#include <hip/hip_bf16.h>

typedef __bf16 bf16;
typedef bf16  v16bf __attribute__((ext_vector_type(16)));
typedef bf16  v8bf  __attribute__((ext_vector_type(8)));
typedef float v8f   __attribute__((ext_vector_type(8)));

static __device__ __forceinline__ float tof(bf16 h) { return (float)h; }
static __device__ __forceinline__ bf16  tobf(float f) { return (bf16)f; }

static __device__ __forceinline__ v16bf combine16(v8bf lo, v8bf hi) {
    v16bf r;
#pragma unroll
    for (int i = 0; i < 8; ++i) { r[i] = lo[i]; r[i + 8] = hi[i]; }
    return r;
}

static __device__ __forceinline__ v8f zero8() {
    v8f z;
#pragma unroll
    for (int i = 0; i < 8; ++i) z[i] = 0.0f;
    return z;
}

// Async global->LDS copy of 16 bytes per lane (CDNA5, ASYNCcnt-tracked).
static __device__ __forceinline__ void async_copy_b128(const bf16* gsrc, bf16* ldst) {
    uint32_t lds_off = (uint32_t)(uintptr_t)ldst;      // addr[31:0] = LDS offset
    uint64_t gaddr   = (uint64_t)(uintptr_t)gsrc;
    asm volatile("global_load_async_to_lds_b128 %0, %1, off"
                 :: "v"(lds_off), "v"(gaddr) : "memory");
}
static __device__ __forceinline__ void wait_async0() {
    asm volatile("s_wait_asynccnt 0x0" ::: "memory");
}

// ---------------------------------------------------------------------------
// fp32 -> bf16 conversion (row-major copy)
// ---------------------------------------------------------------------------
__global__ __launch_bounds__(256) void f32_to_bf16_kernel(
    const float* __restrict__ x, bf16* __restrict__ y, int n) {
    int i = blockIdx.x * 256 + threadIdx.x;
    if (i < n) y[i] = tobf(x[i]);
}

// ---------------------------------------------------------------------------
// Weight pre-transpose: X[K][N] fp32 -> Y[N][K] bf16 (done once per weight,
// so GEMM B-tiles become contiguous b128 async copies, no per-tile transpose).
// ---------------------------------------------------------------------------
__global__ __launch_bounds__(256) void transpose_f32_to_bf16_kernel(
    const float* __restrict__ X, bf16* __restrict__ Y, int Kdim, int Ndim) {
    int idx = blockIdx.x * 256 + threadIdx.x;
    if (idx >= Kdim * Ndim) return;
    int k = idx / Ndim, n = idx - k * Ndim;
    Y[(size_t)n * Kdim + k] = tobf(X[idx]);
}

// ---------------------------------------------------------------------------
// Tiled bf16 GEMM: C[M,N] = A[M,K] * Bt[N,K]^T, fp32 accumulate via WMMA.
// Block: 256 threads (8 waves). Tile: BM=128, BN=128, BK=32, double-buffered
// LDS filled with GLOBAL_LOAD_ASYNC_TO_LDS_B128.
// Wave grid 4(M) x 2(N); each wave owns a 32x64 subtile (2x4 WMMA accums):
// 8 WMMAs per K-step against 12 b128 LDS fragment loads.
// ---------------------------------------------------------------------------
template <bool OUT_BF16>
__global__ __launch_bounds__(256) void gemm_bf16_wmma(
    const bf16* __restrict__ A, const bf16* __restrict__ Bt,
    void* __restrict__ Cout, int M, int N, int K) {
    constexpr int BM = 128, BN = 128, BK = 32;
    __shared__ alignas(16) bf16 As[2][BM * BK];   // [m][k] row-major, 2 x 8 KB
    __shared__ alignas(16) bf16 Bs[2][BN * BK];   // [n][k] row-major, 2 x 8 KB

    const int tid  = threadIdx.x;
    const int wid  = tid >> 5;
    const int lane = tid & 31;
    const int l16  = lane & 15;
    const int lh   = lane >> 4;     // 0 or 1
    const int wm   = wid >> 1;      // 0..3 -> 32 rows each
    const int wn   = wid & 1;       // 0..1 -> 64 cols each
    const int bm   = blockIdx.y * BM;
    const int bn   = blockIdx.x * BN;

    // Per-thread tile-fill coordinates (2 x b128 per thread per tile).
    const int fr = tid >> 2;            // row 0..63 (A) / 0..63 (B)
    const int fc = (tid & 3) * 8;       // col 0,8,16,24

    v8f acc[2][4];
#pragma unroll
    for (int a = 0; a < 2; ++a)
#pragma unroll
        for (int b = 0; b < 4; ++b) acc[a][b] = zero8();

    auto load_tiles = [&](int k0, int buf) {
#pragma unroll
        for (int i = 0; i < 2; ++i) {
            int r = fr + i * 64;
            async_copy_b128(&A[(size_t)(bm + r) * K + k0 + fc],
                            &As[buf][r * BK + fc]);
            async_copy_b128(&Bt[(size_t)(bn + r) * K + k0 + fc],
                            &Bs[buf][r * BK + fc]);
        }
    };

    load_tiles(0, 0);
    wait_async0();
    __syncthreads();

    int cur = 0;
    for (int k0 = 0; k0 < K; k0 += BK) {
        if (k0 + BK < K) load_tiles(k0 + BK, cur ^ 1);

        // Fragments: 2 A frags (rows), 4 B frags (cols), all 2 x ds_load_b128.
        v16bf afrag[2], bfrag[4];
#pragma unroll
        for (int mt = 0; mt < 2; ++mt) {
            const bf16* ap = &As[cur][(wm * 32 + mt * 16 + l16) * BK + lh * 8];
            afrag[mt] = combine16(*(const v8bf*)ap, *(const v8bf*)(ap + 16));
        }
#pragma unroll
        for (int nt = 0; nt < 4; ++nt) {
            const bf16* bp = &Bs[cur][(wn * 64 + nt * 16 + l16) * BK + lh * 8];
            bfrag[nt] = combine16(*(const v8bf*)bp, *(const v8bf*)(bp + 16));
        }
#pragma unroll
        for (int mt = 0; mt < 2; ++mt)
#pragma unroll
            for (int nt = 0; nt < 4; ++nt)
                acc[mt][nt] = __builtin_amdgcn_wmma_f32_16x16x32_bf16(
                    false, afrag[mt], false, bfrag[nt], (short)0,
                    acc[mt][nt], false, false);

        wait_async0();      // my next-tile async copies landed
        __syncthreads();    // everyone done reading cur & writing next
        cur ^= 1;
    }

#pragma unroll
    for (int mt = 0; mt < 2; ++mt)
#pragma unroll
        for (int nt = 0; nt < 4; ++nt)
#pragma unroll
            for (int r = 0; r < 8; ++r) {
                int row = bm + wm * 32 + mt * 16 + lh * 8 + r;
                int col = bn + wn * 64 + nt * 16 + l16;
                if (OUT_BF16)
                    ((bf16*)Cout)[(size_t)row * N + col] = tobf(acc[mt][nt][r]);
                else
                    ((float*)Cout)[(size_t)row * N + col] = acc[mt][nt][r];
            }
}

// ---------------------------------------------------------------------------
// RoPE (interleaved pairs) + head permute:
//   X[B*T][H*64]  ->  Y[(b*H+h)*T + t][64]      (do_rope=0 => permute only)
// ---------------------------------------------------------------------------
__global__ __launch_bounds__(256) void rope_permute_kernel(
    const bf16* __restrict__ X, bf16* __restrict__ Y, int do_rope) {
    int idx = blockIdx.x * 256 + threadIdx.x;    // over B*H*T*32 = 2^21
    int i = idx & 31;
    int t = (idx >> 5) & 2047;
    int h = (idx >> 16) & 15;
    int b = (idx >> 20) & 1;

    const bf16* xp = X + (size_t)(b * 2048 + t) * 1024 + h * 64 + 2 * i;
    float x1 = tof(xp[0]), x2 = tof(xp[1]);
    float y1 = x1, y2 = x2;
    if (do_rope) {
        float freq = __expf(-(float)(2 * i) * (1.0f / 64.0f) * __logf(10000.0f));
        float ang = (float)t * freq;
        float sn, cs;
        __sincosf(ang, &sn, &cs);
        y1 = x1 * cs - x2 * sn;
        y2 = x1 * sn + x2 * cs;
    }
    bf16* yp = Y + ((size_t)(b * 16 + h) * 2048 + t) * 64 + 2 * i;
    yp[0] = tobf(y1);
    yp[1] = tobf(y2);
}

// ---------------------------------------------------------------------------
// Causal flash attention, d_head=64, T=2048.
// Q,K,V: [B*H][T][64] bf16.  Out: [B*T][H*64] bf16.
// Block = 128 threads (4 waves); each wave owns 16 query rows.
// ---------------------------------------------------------------------------
__global__ __launch_bounds__(128) void mla_attn_kernel(
    const bf16* __restrict__ Q, const bf16* __restrict__ Km,
    const bf16* __restrict__ Vm, bf16* __restrict__ Out) {
    constexpr int DH = 64, BKEY = 32;
    const int bh = blockIdx.x;          // b*16 + h
    const int qb = blockIdx.y * 64;     // query tile base
    const int b = bh >> 4;
    const int h = bh & 15;
    const int tid = threadIdx.x;
    const int wid = tid >> 5;
    const int lane = tid & 31;
    const int l16 = lane & 15;
    const int lh = lane >> 4;

    __shared__ alignas(16) bf16 Ks[BKEY * DH];    // [key][d]  (4 KB)
    __shared__ alignas(16) bf16 Vts[DH * BKEY];   // [d][key]  (4 KB)
    __shared__ alignas(16) bf16 Ps[4][16 * 32];   // wave-private P (4 KB)

    const size_t base = (size_t)bh * 2048 * DH;
    const int qrow_frag = qb + wid * 16 + l16;    // row this lane loads (A layout)

    v16bf qa[2];
#pragma unroll
    for (int c = 0; c < 2; ++c) {
        const bf16* qp = Q + base + (size_t)qrow_frag * DH + c * 32 + lh * 8;
        qa[c] = combine16(*(const v8bf*)qp, *(const v8bf*)(qp + 16));
    }

    float mrow[8], lrow[8];
#pragma unroll
    for (int r = 0; r < 8; ++r) { mrow[r] = -1e30f; lrow[r] = 0.0f; }
    v8f o[4];
#pragma unroll
    for (int d = 0; d < 4; ++d) o[d] = zero8();

    const int nkb = (qb + 64) / BKEY;   // causal: only blocks touching keys <= q
    for (int kb = 0; kb < nkb; ++kb) {
        const int kbase = kb * BKEY;
#pragma unroll
        for (int i = 0; i < 2; ++i) {
            int idx = tid + i * 128;
            int r = idx >> 3;
            int c = (idx & 7) * 8;
            *(uint4*)&Ks[r * DH + c] =
                *(const uint4*)&Km[base + (size_t)(kbase + r) * DH + c];
            v8bf vv = *(const v8bf*)&Vm[base + (size_t)(kbase + r) * DH + c];
#pragma unroll
            for (int j = 0; j < 8; ++j) Vts[(c + j) * BKEY + r] = vv[j];
        }
        __syncthreads();

        // S = Q x K^T
        v8f s[2];
#pragma unroll
        for (int ns = 0; ns < 2; ++ns) {
            v8f a = zero8();
#pragma unroll
            for (int c = 0; c < 2; ++c) {
                const bf16* kp = &Ks[(ns * 16 + l16) * DH + c * 32 + lh * 8];
                v16bf bfrag = combine16(*(const v8bf*)kp, *(const v8bf*)(kp + 16));
                a = __builtin_amdgcn_wmma_f32_16x16x32_bf16(
                    false, qa[c], false, bfrag, (short)0, a, false, false);
            }
            s[ns] = a;
        }

        // Scale + causal mask (C layout: row = qb+wid*16+lh*8+r, col = kbase+ns*16+l16)
#pragma unroll
        for (int ns = 0; ns < 2; ++ns) {
            int kg = kbase + ns * 16 + l16;
#pragma unroll
            for (int r = 0; r < 8; ++r) {
                int qg = qb + wid * 16 + lh * 8 + r;
                float v = s[ns][r] * 0.125f;
                s[ns][r] = (kg > qg) ? -1e30f : v;
            }
        }

        // Online softmax per row.
        float alpha[8];
#pragma unroll
        for (int r = 0; r < 8; ++r) {
            float mb = fmaxf(s[0][r], s[1][r]);
#pragma unroll
            for (int off = 8; off >= 1; off >>= 1)
                mb = fmaxf(mb, __shfl_xor(mb, off, 32));
            float mn = fmaxf(mrow[r], mb);
            alpha[r] = __expf(mrow[r] - mn);
            mrow[r] = mn;
            float rs = 0.0f;
#pragma unroll
            for (int ns = 0; ns < 2; ++ns) {
                float p = __expf(s[ns][r] - mn);
                s[ns][r] = p;
                rs += p;
            }
#pragma unroll
            for (int off = 8; off >= 1; off >>= 1)
                rs += __shfl_xor(rs, off, 32);
            lrow[r] = lrow[r] * alpha[r] + rs;
        }
#pragma unroll
        for (int d = 0; d < 4; ++d)
#pragma unroll
            for (int r = 0; r < 8; ++r) o[d][r] *= alpha[r];

        // Re-layout P (C layout) -> A layout via wave-private LDS.
        bf16* pw = Ps[wid];
#pragma unroll
        for (int ns = 0; ns < 2; ++ns)
#pragma unroll
            for (int r = 0; r < 8; ++r)
                pw[(lh * 8 + r) * 32 + ns * 16 + l16] = tobf(s[ns][r]);

        const bf16* pp = &pw[l16 * 32 + lh * 8];
        v16bf pa = combine16(*(const v8bf*)pp, *(const v8bf*)(pp + 16));

        // O += P x V
#pragma unroll
        for (int dt = 0; dt < 4; ++dt) {
            const bf16* vp = &Vts[(dt * 16 + l16) * BKEY + lh * 8];
            v16bf vb = combine16(*(const v8bf*)vp, *(const v8bf*)(vp + 16));
            o[dt] = __builtin_amdgcn_wmma_f32_16x16x32_bf16(
                false, pa, false, vb, (short)0, o[dt], false, false);
        }
        __syncthreads();
    }

#pragma unroll
    for (int r = 0; r < 8; ++r) {
        float inv = 1.0f / lrow[r];
        int row = b * 2048 + qb + wid * 16 + lh * 8 + r;
#pragma unroll
        for (int dt = 0; dt < 4; ++dt) {
            int col = h * 64 + dt * 16 + l16;
            Out[(size_t)row * 1024 + col] = tobf(o[dt][r] * inv);
        }
    }
}

// ---------------------------------------------------------------------------
// Host-side launch
// ---------------------------------------------------------------------------
extern "C" void kernel_launch(void* const* d_in, const int* in_sizes, int n_in,
                              void* d_out, int out_size, void* d_ws, size_t ws_size,
                              hipStream_t stream) {
    (void)in_sizes; (void)n_in; (void)out_size; (void)ws_size;
    const float* q   = (const float*)d_in[0];
    const float* k   = (const float*)d_in[1];
    const float* v   = (const float*)d_in[2];
    const float* Wq  = (const float*)d_in[3];
    const float* Wkc = (const float*)d_in[4];
    const float* Wvc = (const float*)d_in[5];
    const float* Wk  = (const float*)d_in[6];
    const float* Wv  = (const float*)d_in[7];
    const float* Wo  = (const float*)d_in[8];

    constexpr size_t NQ = 4096ull * 1024;   // tokens x d_model
    constexpr size_t NC = 4096ull * 256;
    bf16* w = (bf16*)d_ws;
    size_t off = 0;
    bf16* qb   = w + off; off += NQ;
    bf16* kb   = w + off; off += NQ;
    bf16* vb   = w + off; off += NQ;
    bf16* WqT  = w + off; off += 1024ull * 1024;  // [N=1024][K=1024]
    bf16* WkcT = w + off; off += 1024ull * 256;   // [N=256][K=1024]
    bf16* WvcT = w + off; off += 1024ull * 256;
    bf16* WkT  = w + off; off += 256ull * 1024;   // [N=1024][K=256]
    bf16* WvT  = w + off; off += 256ull * 1024;
    bf16* WoT  = w + off; off += 1024ull * 1024;
    bf16* Qp   = w + off; off += NQ;
    bf16* Kc   = w + off; off += NC;
    bf16* Kp   = w + off; off += NQ;
    bf16* Vc   = w + off; off += NC;
    bf16* Vp   = w + off; off += NQ;
    bf16* Qr   = w + off; off += NQ;
    bf16* Kr   = w + off; off += NQ;
    bf16* Vr   = w + off; off += NQ;
    bf16* Ao   = w + off; off += NQ;

    auto cvt = [&](const float* x, bf16* y, size_t n) {
        f32_to_bf16_kernel<<<(unsigned)((n + 255) / 256), 256, 0, stream>>>(x, y, (int)n);
    };
    auto cvtT = [&](const float* x, bf16* y, int Kd, int Nd) {
        size_t n = (size_t)Kd * Nd;
        transpose_f32_to_bf16_kernel<<<(unsigned)((n + 255) / 256), 256, 0, stream>>>(x, y, Kd, Nd);
    };
    cvt(q, qb, NQ);
    cvt(k, kb, NQ);
    cvt(v, vb, NQ);
    cvtT(Wq,  WqT,  1024, 1024);
    cvtT(Wkc, WkcT, 1024, 256);
    cvtT(Wvc, WvcT, 1024, 256);
    cvtT(Wk,  WkT,  256, 1024);
    cvtT(Wv,  WvT,  256, 1024);
    cvtT(Wo,  WoT,  1024, 1024);

    dim3 blk(256);
    // Q = q @ Wq                         (4096 x 1024 x 1024)
    gemm_bf16_wmma<true><<<dim3(1024 / 128, 4096 / 128), blk, 0, stream>>>(qb, WqT, Qp, 4096, 1024, 1024);
    // Kc = k @ Wk_c ; K = Kc @ Wk
    gemm_bf16_wmma<true><<<dim3(256 / 128, 4096 / 128), blk, 0, stream>>>(kb, WkcT, Kc, 4096, 256, 1024);
    gemm_bf16_wmma<true><<<dim3(1024 / 128, 4096 / 128), blk, 0, stream>>>(Kc, WkT, Kp, 4096, 1024, 256);
    // Vc = v @ Wv_c ; V = Vc @ Wv
    gemm_bf16_wmma<true><<<dim3(256 / 128, 4096 / 128), blk, 0, stream>>>(vb, WvcT, Vc, 4096, 256, 1024);
    gemm_bf16_wmma<true><<<dim3(1024 / 128, 4096 / 128), blk, 0, stream>>>(Vc, WvT, Vp, 4096, 1024, 256);

    // RoPE + head permute (V: permute only)
    rope_permute_kernel<<<8192, 256, 0, stream>>>(Qp, Qr, 1);
    rope_permute_kernel<<<8192, 256, 0, stream>>>(Kp, Kr, 1);
    rope_permute_kernel<<<8192, 256, 0, stream>>>(Vp, Vr, 0);

    // Causal flash attention: grid = (B*H, T/64)
    mla_attn_kernel<<<dim3(32, 32), 128, 0, stream>>>(Qr, Kr, Vr, Ao);

    // Final projection: out = attn @ Wo (fp32 output)
    gemm_bf16_wmma<false><<<dim3(1024 / 128, 4096 / 128), blk, 0, stream>>>(Ao, WoT, d_out, 4096, 1024, 1024);
}
